// JCF_FFT_MLP_34041910788889
// MI455X (gfx1250) — matrix operand order repaired and checked by reading, not compiled, not run on hardware
//
#include <hip/hip_runtime.h>
#include <hip/hip_bf16.h>
#include <math.h>

typedef __attribute__((ext_vector_type(16))) __bf16 v16bf;
typedef __attribute__((ext_vector_type(8)))  __bf16 v8bf;
typedef __attribute__((ext_vector_type(8)))  float  v8f;

#define TOK        16384
#define TPBATCH    4096
#define NF         60
#define WIN        64
#define NBINS      33
#define TOTAL_IN   2040
#define K1PAD      2048
#define HID        256
#define H3DIM      128
#define NOUT       3

// 5 N-tiles (80 cols: interleaved re/im pairs for 33 bins, padded) x 2 K-tiles
#define DFT_NT     5
#define DFT_KT     2

// workspace layout (bytes, all 256-aligned)
#define OFF_DFT    ((size_t)0)                       // 10 tiles * 1024B
#define OFF_W1T    ((size_t)10240)                   // 64*16 tiles * 1024B = 1 MB
#define OFF_W2T    (OFF_W1T + (size_t)64*16*1024)    // 8*16 tiles
#define OFF_W3T    (OFF_W2T + (size_t)8*16*1024)     // 8*8 tiles
#define OFF_H0     (OFF_W3T + (size_t)8*8*1024)      // 16384*2048 bf16 = 64 MB
#define OFF_H1     (OFF_H0  + (size_t)TOK*K1PAD*2)   // 16384*256 bf16
#define OFF_H2     (OFF_H1  + (size_t)TOK*HID*2)
#define OFF_H3     (OFF_H2  + (size_t)TOK*HID*2)     // 16384*128 f32

// Fast epilogue math: result is rounded to bf16 (8-bit mantissa), so the
// hardware v_sqrt_f32 / v_log_f32 (log2) are plenty accurate.
__device__ __forceinline__ float fast_log1p_sqrt(float s2) {
    float mag = __builtin_amdgcn_sqrtf(s2);                  // v_sqrt_f32
    return 0.69314718056f * __builtin_amdgcn_logf(1.0f + mag); // v_log_f32 * ln2
}

// ---------------------------------------------------------------------------
// Prep: DFT basis tiles in WMMA-B swizzle. Column 2b = cos(2*pi*b*k/64),
// column 2b+1 = -sin(2*pi*b*k/64); columns >= 66 are zero padding.
// B-layout: lane L -> col nt*16+(L&15); element e -> k = kt*32 + (L<16?0:8)+e+(e>=8?8:0).
// ---------------------------------------------------------------------------
__global__ void prep_dft_kernel(__bf16* __restrict__ tiles) {
    int idx = blockIdx.x * blockDim.x + threadIdx.x;
    if (idx >= DFT_NT * DFT_KT * 512) return;
    int tile   = idx >> 9;
    int within = idx & 511;
    int nt = tile >> 1, kh = tile & 1;
    int L = within >> 4, e = within & 15;
    int n = nt * 16 + (L & 15);
    int k = kh * 32 + ((L < 16) ? 0 : 8) + e + ((e >= 8) ? 8 : 0);
    float v = 0.0f;
    if (n < 2 * NBINS) {
        int   b   = n >> 1;
        float ang = (6.283185307179586f / (float)WIN) * (float)(b * k);
        v = ((n & 1) == 0) ? cosf(ang) : -sinf(ang);
    }
    tiles[idx] = (__bf16)v;
}

// ---------------------------------------------------------------------------
// Prep: fp32 weight [Kreal x N] row-major -> bf16 WMMA-B swizzled tiles,
// K zero-padded up to KT*32.
// ---------------------------------------------------------------------------
__global__ void prep_w_kernel(const float* __restrict__ W, __bf16* __restrict__ tiles,
                              int Kreal, int N, int NT, int totalHalfs) {
    int idx = blockIdx.x * blockDim.x + threadIdx.x;
    if (idx >= totalHalfs) return;
    int tile   = idx >> 9;
    int within = idx & 511;
    int kt = tile / NT, nt = tile % NT;
    int L = within >> 4, e = within & 15;
    int n = nt * 16 + (L & 15);
    int k = kt * 32 + ((L < 16) ? 0 : 8) + e + ((e >= 8) ? 8 : 0);
    float v = (k < Kreal) ? W[(size_t)k * N + n] : 0.0f;
    tiles[idx] = (__bf16)v;
}

// ---------------------------------------------------------------------------
// FFT feature kernel: one block = 16 tokens (never straddles a batch).
// LDS holds the reflect-padded 79x60 f32 time slab. 8 waves; wave w owns
// features f = w, w+8, ... Each feature: A(16x64 bf16) x DFT(64x80 bf16)
// via 10 v_wmma_f32_16x16x32_bf16, then re/im pairing via shfl_xor(1),
// log1p(sqrt(re^2+im^2)) -> bf16 h0.
// ---------------------------------------------------------------------------
__global__ void __launch_bounds__(256)
fft_feat_kernel(const float* __restrict__ x, const __bf16* __restrict__ dftTiles,
                __bf16* __restrict__ h0) {
    __shared__ float sx[79 * NF];

    const int tid  = threadIdx.x;
    const int lane = tid & 31;
    const int wave = tid >> 5;
    const int t0g  = blockIdx.x * 16;        // global token base
    const int bat  = t0g / TPBATCH;
    const int t0   = t0g % TPBATCH;

    // Fill reflect-padded slab: slab s=0 corresponds to time t0-32.
    for (int idx = tid; idx < 79 * NF; idx += 256) {
        int s = idx / NF, f = idx % NF;
        int tt = t0 - 32 + s;
        if (tt < 0) tt = -tt;
        if (tt > TPBATCH - 1) tt = 2 * (TPBATCH - 1) - tt;
        sx[idx] = x[((size_t)bat * TPBATCH + tt) * NF + f];
    }
    __syncthreads();

    // Raw features + zero pad of h0 row.
    if (tid < 16 * NF) {
        int i = tid / NF, f = tid % NF;
        h0[(size_t)(t0g + i) * K1PAD + f] = (__bf16)sx[(32 + i) * NF + f];
    }
    if (tid < 16 * (K1PAD - TOTAL_IN)) {
        int i = tid / (K1PAD - TOTAL_IN), j = tid % (K1PAD - TOTAL_IN);
        h0[(size_t)(t0g + i) * K1PAD + TOTAL_IN + j] = (__bf16)0.0f;
    }

    // DFT basis B-operands resident in registers (same for all waves).
    v16bf Bt[DFT_NT * DFT_KT];
#pragma unroll
    for (int t = 0; t < DFT_NT * DFT_KT; ++t)
        Bt[t] = *(const v16bf*)(dftTiles + (size_t)t * 512 + lane * 16);

    const int m    = lane & 15;
    const int base = (lane < 16) ? 0 : 8;
    const int rofs = (lane < 16) ? 0 : 8;     // C-row offset for upper half-wave
    const bool isRe = ((m & 1) == 0);

    for (int f = wave; f < NF; f += 8) {
        // A operands per documented 16-bit A layout:
        // e<8  -> k = base+e ;  e>=8 -> k = 16+base+(e-8)
        v16bf a0, a1;
#pragma unroll
        for (int e = 0; e < 16; ++e) {
            int k0 = base + e + ((e >= 8) ? 8 : 0);
            a0[e] = (__bf16)sx[(m + k0) * NF + f];
            a1[e] = (__bf16)sx[(m + 32 + k0) * NF + f];
        }
#pragma unroll
        for (int nt = 0; nt < DFT_NT; ++nt) {
            v8f c = {};
            c = __builtin_amdgcn_wmma_f32_16x16x32_bf16(false, a0, false, Bt[nt * 2 + 0],
                                                        (short)0, c, false, false);
            c = __builtin_amdgcn_wmma_f32_16x16x32_bf16(false, a1, false, Bt[nt * 2 + 1],
                                                        (short)0, c, false, false);
            int ncol = nt * 16 + m;           // 0..79 ; pairs (2b, 2b+1) = (re, im)
            int bin  = ncol >> 1;
#pragma unroll
            for (int r = 0; r < 8; ++r) {
                float v   = c[r];
                float o   = __shfl_xor(v, 1, 32);
                float re  = isRe ? v : o;
                float im  = isRe ? o : v;
                float mag = fast_log1p_sqrt(re * re + im * im);
                if (isRe && bin < NBINS) {
                    int tok = t0g + r + rofs;
                    h0[(size_t)tok * K1PAD + NF + f * NBINS + bin] = (__bf16)mag;
                }
            }
        }
    }
}

// ---------------------------------------------------------------------------
// Generic WMMA MLP layer: out[16 x N] per block, 8 waves x TPW N-tiles.
// A from row-major bf16 activations (two 16B loads/lane/k-step),
// B from pre-swizzled bf16 weight tiles (one 32B load/lane/tile).
// global_prefetch_b8 hints cover the next k-step's A/B cachelines.
// ---------------------------------------------------------------------------
template <int K, int N, int TPW, bool RELU, bool OUT_BF16>
__global__ void __launch_bounds__(256)
mlp_layer_kernel(const __bf16* __restrict__ Ain, int strideA,
                 const __bf16* __restrict__ Wt, const float* __restrict__ bias,
                 void* __restrict__ Out, int strideO) {
    constexpr int KT = K / 32;
    constexpr int NT = N / 16;

    const int lane = threadIdx.x & 31;
    const int wave = threadIdx.x >> 5;
    const int t0   = blockIdx.x * 16;
    const int m    = lane & 15;
    const int base = (lane < 16) ? 0 : 8;
    const int rofs = (lane < 16) ? 0 : 8;

    const __bf16* arow = Ain + (size_t)(t0 + m) * strideA;

    v8f acc[TPW];
#pragma unroll
    for (int j = 0; j < TPW; ++j) acc[j] = (v8f){};

    for (int kt = 0; kt < KT; ++kt) {
        if (kt + 1 < KT) {
            __builtin_prefetch(arow + (kt + 1) * 32 + base, 0, 1);
            __builtin_prefetch(Wt + ((size_t)((kt + 1) * NT + wave * TPW) * 512) + lane * 16, 0, 1);
        }
        v8bf c0 = *(const v8bf*)(arow + kt * 32 + base);
        v8bf c1 = *(const v8bf*)(arow + kt * 32 + 16 + base);
        v16bf a = __builtin_shufflevector(c0, c1, 0, 1, 2, 3, 4, 5, 6, 7,
                                          8, 9, 10, 11, 12, 13, 14, 15);
#pragma unroll
        for (int j = 0; j < TPW; ++j) {
            int nt = wave * TPW + j;
            v16bf b = *(const v16bf*)(Wt + ((size_t)(kt * NT + nt) * 512) + lane * 16);
            acc[j] = __builtin_amdgcn_wmma_f32_16x16x32_bf16(false, a, false, b,
                                                             (short)0, acc[j], false, false);
        }
    }

#pragma unroll
    for (int j = 0; j < TPW; ++j) {
        int   nt = wave * TPW + j;
        int   n  = nt * 16 + m;
        float bv = bias[n];
#pragma unroll
        for (int r = 0; r < 8; ++r) {
            int   tok = t0 + r + rofs;
            float v   = acc[j][r] + bv;
            if (RELU) v = fmaxf(v, 0.0f);
            if (OUT_BF16)
                ((__bf16*)Out)[(size_t)tok * strideO + n] = (__bf16)v;
            else
                ((float*)Out)[(size_t)tok * strideO + n] = v;
        }
    }
}

// ---------------------------------------------------------------------------
// Tiny final layer: 128 -> 3, fp32 VALU, one token per thread.
// ---------------------------------------------------------------------------
__global__ void layer4_kernel(const float* __restrict__ h3, const float* __restrict__ W4,
                              const float* __restrict__ b4, float* __restrict__ out) {
    int tok = blockIdx.x * blockDim.x + threadIdx.x;
    if (tok >= TOK) return;
    const float* hr = h3 + (size_t)tok * H3DIM;
    float a0 = b4[0], a1 = b4[1], a2 = b4[2];
#pragma unroll 4
    for (int i = 0; i < H3DIM; ++i) {
        float h = hr[i];
        a0 += h * W4[i * 3 + 0];
        a1 += h * W4[i * 3 + 1];
        a2 += h * W4[i * 3 + 2];
    }
    out[(size_t)tok * 3 + 0] = a0;
    out[(size_t)tok * 3 + 1] = a1;
    out[(size_t)tok * 3 + 2] = a2;
}

// ---------------------------------------------------------------------------
extern "C" void kernel_launch(void* const* d_in, const int* in_sizes, int n_in,
                              void* d_out, int out_size, void* d_ws, size_t ws_size,
                              hipStream_t stream) {
    (void)in_sizes; (void)n_in; (void)out_size; (void)ws_size;
    const float* x  = (const float*)d_in[0];
    const float* W1 = (const float*)d_in[1];
    const float* b1 = (const float*)d_in[2];
    const float* W2 = (const float*)d_in[3];
    const float* b2 = (const float*)d_in[4];
    const float* W3 = (const float*)d_in[5];
    const float* b3 = (const float*)d_in[6];
    const float* W4 = (const float*)d_in[7];
    const float* b4 = (const float*)d_in[8];
    float* out = (float*)d_out;

    char* ws = (char*)d_ws;
    __bf16* dftT = (__bf16*)(ws + OFF_DFT);
    __bf16* W1T  = (__bf16*)(ws + OFF_W1T);
    __bf16* W2T  = (__bf16*)(ws + OFF_W2T);
    __bf16* W3T  = (__bf16*)(ws + OFF_W3T);
    __bf16* h0   = (__bf16*)(ws + OFF_H0);
    __bf16* h1   = (__bf16*)(ws + OFF_H1);
    __bf16* h2   = (__bf16*)(ws + OFF_H2);
    float*  h3   = (float*)(ws + OFF_H3);

    // Prep (stream-ordered, deterministic every call)
    prep_dft_kernel<<<(DFT_NT * DFT_KT * 512 + 255) / 256, 256, 0, stream>>>(dftT);
    {
        int th = 64 * 16 * 512;   // W1: K=2048 (64 kt), N=256 (16 nt)
        prep_w_kernel<<<(th + 255) / 256, 256, 0, stream>>>(W1, W1T, TOTAL_IN, HID, 16, th);
    }
    {
        int th = 8 * 16 * 512;    // W2: K=256, N=256
        prep_w_kernel<<<(th + 255) / 256, 256, 0, stream>>>(W2, W2T, HID, HID, 16, th);
    }
    {
        int th = 8 * 8 * 512;     // W3: K=256, N=128
        prep_w_kernel<<<(th + 255) / 256, 256, 0, stream>>>(W3, W3T, HID, H3DIM, 8, th);
    }

    // FFT features + raw concat -> h0 [16384 x 2048] bf16
    fft_feat_kernel<<<TOK / 16, 256, 0, stream>>>(x, dftT, h0);

    // MLP layers (WMMA bf16, f32 accumulate)
    mlp_layer_kernel<K1PAD, HID, 2, true, true>
        <<<TOK / 16, 256, 0, stream>>>(h0, K1PAD, W1T, b1, (void*)h1, HID);
    mlp_layer_kernel<HID, HID, 2, true, true>
        <<<TOK / 16, 256, 0, stream>>>(h1, HID, W2T, b2, (void*)h2, HID);
    mlp_layer_kernel<HID, H3DIM, 1, true, false>
        <<<TOK / 16, 256, 0, stream>>>(h2, HID, W3T, b3, (void*)h3, H3DIM);

    layer4_kernel<<<(TOK + 255) / 256, 256, 0, stream>>>(h3, W4, b4, out);
}
